// CifMiddleware_73220602462363
// MI455X (gfx1250) — matrix operand order repaired and checked by reading, not compiled, not run on hardware
//
#include <hip/hip_runtime.h>
#include <hip/hip_bf16.h>

typedef __attribute__((ext_vector_type(16))) _Float16 v16h;
typedef __attribute__((ext_vector_type(8)))  _Float16 v8h;
typedef __attribute__((ext_vector_type(8)))  float    v8f;

namespace {
constexpr int B_   = 16;
constexpr int T_   = 2048;
constexpr int C_   = 1024;   // encoder channels == GEMM K
constexpr int DU_  = 1024;   // dense units == GEMM1 N
constexpr int OUT_ = 512;    // output proj N
constexpr int KMAX = 1024;   // compacted fired-frame capacity (n_fired <= target_len+1 <= ~501)
constexpr int ROWS = 64;     // M rows per workgroup in GEMM1 (LDS A panel = 64*1024*2 = 128KB)
}

// Build a 16-half A fragment from two contiguous 8-half groups.
__device__ __forceinline__ v16h make_frag(const _Float16* p, int gap) {
  v8h lo = *(const v8h*)p;
  v8h hi = *(const v8h*)(p + gap);
  v16h a;
#pragma unroll
  for (int i = 0; i < 8; ++i) { a[i] = lo[i]; a[8 + i] = hi[i]; }
  return a;
}

// ---------------------------------------------------------------- K0: casts + init
__global__ void cif_cvt_weights(const float* __restrict__ Wd, const float* __restrict__ Wo,
                                _Float16* __restrict__ wdH, _Float16* __restrict__ woH,
                                float* __restrict__ wsum) {
  const int tid = blockIdx.x * blockDim.x + threadIdx.x;
  const int stride = gridDim.x * blockDim.x;
  for (int i = tid; i < DU_ * C_; i += stride) wdH[i] = (_Float16)Wd[i];
  for (int i = tid; i < OUT_ * C_; i += stride) woH[i] = (_Float16)Wo[i];
  if (tid < B_) wsum[tid] = 0.0f;
}

// ---------------------------------------------------------------- K0b: zero cif_outputs
__global__ void cif_zero(float* __restrict__ p, size_t n) {
  size_t i = ((size_t)blockIdx.x * blockDim.x + threadIdx.x) * 4;
  const size_t stride = (size_t)gridDim.x * blockDim.x * 4;
  for (; i + 3 < n; i += stride) {
    float4 z; z.x = 0.f; z.y = 0.f; z.z = 0.f; z.w = 0.f;
    *(float4*)(p + i) = z;
  }
}

// ---------------------------------------------------------------- K1: fused GEMM1 -> weight
// org_weight[m] = not_pad * sigmoid( sum_d relu( X[m,:]·Wd[d,:] + bd[d] ) * Wp[d] + bp )
// Also accumulates per-batch sum into wsum[b].
__global__ __launch_bounds__(256)
void cif_gemm1_weight(const float* __restrict__ enc, const _Float16* __restrict__ wdH,
                      const float* __restrict__ bd, const float* __restrict__ Wp,
                      const float* __restrict__ bp, const int* __restrict__ inLen,
                      float* __restrict__ orgW, float* __restrict__ wsum) {
  extern __shared__ char smem[];
  _Float16* aPanel = (_Float16*)smem;                 // [ROWS][C_] f16
  float*    rowSum = (float*)(smem + ROWS * C_ * 2);  // [ROWS]

  const int mBase = blockIdx.x * ROWS;
  const int tid   = threadIdx.x;
  if (tid < ROWS) rowSum[tid] = 0.0f;

  // Cooperative load+cast: 64 rows x 1024 f32 -> f16 LDS panel.
  const float* src = enc + (size_t)mBase * C_;
  for (int i = tid * 4; i < ROWS * C_; i += 256 * 4) {
    float4 v = *(const float4*)(src + i);
    _Float16* d = aPanel + i;
    d[0] = (_Float16)v.x; d[1] = (_Float16)v.y; d[2] = (_Float16)v.z; d[3] = (_Float16)v.w;
  }
  __syncthreads();

  const int wave   = tid >> 5;
  const int lane   = tid & 31;
  const int laneHi = lane >> 4;   // 0: lanes 0-15, 1: lanes 16-31
  const int lane15 = lane & 15;

  // Deferred d-reduction: accumulate relu(h+bd)*Wp per (row-tile, row) across all
  // 8 n-tiles in registers; one cross-lane reduce at the end (8x fewer ds_bpermute).
  float gAcc[4][8];
#pragma unroll
  for (int rb = 0; rb < 4; ++rb)
#pragma unroll
    for (int i = 0; i < 8; ++i) gAcc[rb][i] = 0.0f;

  // 8 waves x 8 n-tiles = 64 tiles of 16 -> full DU_=1024.
  for (int ntSub = 0; ntSub < 8; ++ntSub) {
    const int dBase = (wave * 8 + ntSub) * 16;
    v8f acc[4];
#pragma unroll
    for (int rb = 0; rb < 4; ++rb) acc[rb] = {};

    // B 32x16 frag: lane<16 -> col dBase+lane, K=k..k+15; lane>=16 -> K=k+16..k+31.
    // Explicit double buffer (bcur/bmid): loads land directly in the consuming
    // registers two half-steps later -> no register rotation, no hazard NOPs.
    const _Float16* bRow = wdH + (size_t)(dBase + lane15) * C_ + laneHi * 16;
    v16h bcur = *(const v16h*)bRow;  // k = 0
#pragma unroll 1
    for (int k = 0; k < C_; k += 64) {
      v16h bmid = *(const v16h*)(bRow + k + 32);
#pragma unroll
      for (int rb = 0; rb < 4; ++rb) {
        // A 16x32 frag from LDS: row rb*16+lane15; K {0..15} / {16..31}.
        const _Float16* app = aPanel + (rb * 16 + lane15) * C_ + k + laneHi * 16;
        acc[rb] = __builtin_amdgcn_wmma_f32_16x16x32_f16(
            false, make_frag(app, 8), false, bcur, (short)0, acc[rb], false, false);
      }
      const int kn = (k + 64 < C_) ? (k + 64) : k;  // uniform tail guard, no overread
      bcur = *(const v16h*)(bRow + kn);
#pragma unroll
      for (int rb = 0; rb < 4; ++rb) {
        const _Float16* app = aPanel + (rb * 16 + lane15) * C_ + (k + 32) + laneHi * 16;
        acc[rb] = __builtin_amdgcn_wmma_f32_16x16x32_f16(
            false, make_frag(app, 8), false, bmid, (short)0, acc[rb], false, false);
      }
    }

    // Per-lane epilogue for this n-tile: g = relu(h + bd[d]) * Wp[d], deferred sum.
    const int d = dBase + lane15;
    const float bdv = bd[d];
    const float wpv = Wp[d];
#pragma unroll
    for (int rb = 0; rb < 4; ++rb)
#pragma unroll
      for (int i = 0; i < 8; ++i) {
        float g = acc[rb][i] + bdv;
        gAcc[rb][i] += (g > 0.0f) ? g * wpv : 0.0f;
      }
  }

  // One cross-lane (16-wide) reduction per (rb, row); lanes 0/16 commit to LDS.
#pragma unroll
  for (int rb = 0; rb < 4; ++rb)
#pragma unroll
    for (int i = 0; i < 8; ++i) {
      float g = gAcc[rb][i];
      g += __shfl_xor(g, 1, 16);
      g += __shfl_xor(g, 2, 16);
      g += __shfl_xor(g, 4, 16);
      g += __shfl_xor(g, 8, 16);
      if (lane15 == 0)  // lane 0 -> rows i, lane 16 -> rows 8+i
        atomicAdd(&rowSum[rb * 16 + laneHi * 8 + i], g);
    }
  __syncthreads();

  // Finalize rows: sigmoid + not-pad; stage per-row org weight for batch sum.
  if (tid < ROWS) {
    const int m = mBase + tid;
    const int b = m / T_;
    const int t = m % T_;
    const float val = rowSum[tid] + bp[0];
    const float sg  = 1.0f / (1.0f + __expf(-val));
    const float org = (t < inLen[b]) ? sg : 0.0f;
    orgW[m] = org;
    rowSum[tid] = org;
  }
  __syncthreads();
  if (tid == 0) {
    float s = 0.0f;
#pragma unroll 8
    for (int i = 0; i < ROWS; ++i) s += rowSum[i];
    atomicAdd(&wsum[mBase / T_], s);  // all ROWS rows share one batch (64 | T)
  }
}

// ---------------------------------------------------------------- K2: sequential CIF scan
// One block per batch, 1024 threads = one channel each. Carry is uniform per thread.
__global__ __launch_bounds__(1024)
void cif_scan(const float* __restrict__ enc, const float* __restrict__ orgW,
              const float* __restrict__ wsum, const int* __restrict__ inLen,
              const int* __restrict__ tgtLen, _Float16* __restrict__ compact,
              int* __restrict__ nFired, float* __restrict__ maskOut,
              float* __restrict__ durOut, float* __restrict__ qtyOut) {
  __shared__ int sIdx[KMAX];
  const int b   = blockIdx.x;
  const int tid = threadIdx.x;
  const float scale = (float)tgtLen[b] / (wsum[b] + 1e-8f);
  const int   il    = inLen[b];
  const float* e    = enc + (size_t)b * T_ * C_ + tid;
  const float* wrow = orgW + b * T_;

  float accW = 0.0f, accS = 0.0f;
  int k = 0;
  // One-step register prefetch + deep cache prefetch: the carry chain is serial,
  // so keep encoder rows flowing ahead of it (global_prefetch_b8).
  float wNext = wrow[0] * scale;
  float eNext = e[0];
#pragma unroll 4
  for (int p = 1; p < 8; ++p) __builtin_prefetch(e + (size_t)p * C_, 0, 3);
  for (int t = 0; t < T_; ++t) {
    const float w  = wNext;
    const float ev = eNext;
    if (t + 1 < T_) {
      wNext = wrow[t + 1] * scale;
      eNext = e[(size_t)(t + 1) * C_];
    }
    if (t + 8 < T_) __builtin_prefetch(e + (size_t)(t + 8) * C_, 0, 3);
    const bool fired = (accW + w) >= 1.0f;   // CIF_THRESHOLD
    const float rem  = 1.0f - accW;
    if (fired && (t <= il)) {                // reference zeroes f_s when t > padding_start
      if (k < KMAX) {
        compact[((size_t)b * KMAX + k) * C_ + tid] = (_Float16)(accS + rem * ev);
        if (tid == 0) sIdx[k] = t;
      }
      ++k;
    }
    accS = fired ? (w - rem) * ev : accS + w * ev;
    accW = fired ? (w - rem)      : accW + w;
  }
  if (tid == 0) { nFired[b] = k; qtyOut[b] = wsum[b]; }
  __syncthreads();

  // Padding mask + durations (int outputs written as float values).
  const int kc = (k < KMAX) ? k : KMAX;
  for (int t = tid; t < T_; t += 1024) {
    float mk = (k > 0) ? ((t < k) ? 1.0f : 0.0f) : ((t == 0) ? 1.0f : 0.0f);
    maskOut[b * T_ + t] = mk;
    float du = 0.0f;
    if (t < kc) {
      const int cur  = sIdx[t];
      const int prev = (t > 0) ? sIdx[t - 1] : 0;
      du = (float)(cur - prev);
    }
    durOut[b * T_ + t] = du;
  }
}

// ---------------------------------------------------------------- K3: projection GEMM (masked)
__global__ __launch_bounds__(256)
void cif_proj(const _Float16* __restrict__ compact, const _Float16* __restrict__ woH,
              const int* __restrict__ nFired, float* __restrict__ outC) {
  const int b = blockIdx.y;
  const int n = nFired[b];
  const int rowBase = blockIdx.x * 16;
  if (rowBase >= n) return;  // early-exit: only fired rows are projected

  const int tid = threadIdx.x;
  const int wave = tid >> 5, lane = tid & 31;
  const int laneHi = lane >> 4, lane15 = lane & 15;

  const _Float16* aRow = compact + ((size_t)b * KMAX + rowBase + lane15) * C_ + laneHi * 16;

  for (int nt = 0; nt < 4; ++nt) {  // 8 waves x 4 n-tiles = 32 tiles -> OUT_=512
    const int oBase = (wave * 4 + nt) * 16;
    const _Float16* bRow = woH + (size_t)(oBase + lane15) * C_ + laneHi * 16;
    v8f acc = {};
    v16h bcur = *(const v16h*)bRow;  // k = 0; double-buffered below
#pragma unroll 1
    for (int k = 0; k < C_; k += 64) {
      v16h bmid = *(const v16h*)(bRow + k + 32);
      acc = __builtin_amdgcn_wmma_f32_16x16x32_f16(
          false, make_frag(aRow + k, 8), false, bcur, (short)0, acc, false, false);
      const int kn = (k + 64 < C_) ? (k + 64) : k;
      bcur = *(const v16h*)(bRow + kn);
      acc = __builtin_amdgcn_wmma_f32_16x16x32_f16(
          false, make_frag(aRow + k + 32, 8), false, bmid, (short)0, acc, false, false);
    }
#pragma unroll
    for (int i = 0; i < 8; ++i) {
      const int r = rowBase + laneHi * 8 + i;
      if (r < n)  // rows >= n stay zero (pre-zeroed); stale scratch rows never escape
        outC[((size_t)b * T_ + r) * OUT_ + oBase + lane15] = acc[i];
    }
  }
}

// ---------------------------------------------------------------- launch
extern "C" void kernel_launch(void* const* d_in, const int* in_sizes, int n_in,
                              void* d_out, int out_size, void* d_ws, size_t ws_size,
                              hipStream_t stream) {
  const float* enc    = (const float*)d_in[0];
  const int*   inLen  = (const int*)d_in[1];
  const int*   tgtLen = (const int*)d_in[2];
  const float* Wd     = (const float*)d_in[3];
  const float* bd     = (const float*)d_in[4];
  const float* Wp     = (const float*)d_in[5];
  const float* bp     = (const float*)d_in[6];
  const float* Wo     = (const float*)d_in[7];

  // Output regions (concatenated in return order, float dtype).
  float* outC  = (float*)d_out;                         // [B,T,OUT]
  float* outMk = outC + (size_t)B_ * T_ * OUT_;         // [B,T]
  float* outDu = outMk + (size_t)B_ * T_;               // [B,T]
  float* outQt = outDu + (size_t)B_ * T_;               // [B]

  // Workspace partition (~36.8 MB).
  char* w = (char*)d_ws;
  _Float16* wdH     = (_Float16*)w; w += (size_t)DU_ * C_ * 2;
  _Float16* woH     = (_Float16*)w; w += (size_t)OUT_ * C_ * 2;
  _Float16* compact = (_Float16*)w; w += (size_t)B_ * KMAX * C_ * 2;
  float*    orgW    = (float*)w;    w += (size_t)B_ * T_ * 4;
  float*    wsum    = (float*)w;    w += 64;
  int*      nFired  = (int*)w;      w += 64;

  cif_cvt_weights<<<512, 256, 0, stream>>>(Wd, Wo, wdH, woH, wsum);
  cif_zero<<<2048, 256, 0, stream>>>(outC, (size_t)B_ * T_ * OUT_);

  const int smemBytes = ROWS * C_ * 2 + ROWS * 4;  // 128KB A-panel + row sums
  cif_gemm1_weight<<<(B_ * T_) / ROWS, 256, smemBytes, stream>>>(
      enc, wdH, bd, Wp, bp, inLen, orgW, wsum);

  cif_scan<<<B_, 1024, 0, stream>>>(enc, orgW, wsum, inLen, tgtLen,
                                    compact, nFired, outMk, outDu, outQt);

  cif_proj<<<dim3(KMAX / 16, B_), 256, 0, stream>>>(compact, woH, nFired, outC);
}
